// T2V_FullAttention_11965778886681
// MI455X (gfx1250) — matrix operand order repaired and checked by reading, not compiled, not run on hardware
//
#include <hip/hip_runtime.h>

#define DEVINL __device__ __forceinline__
#define AS1 __attribute__((address_space(1)))
#define AS3 __attribute__((address_space(3)))

typedef __bf16 bf16;
typedef __attribute__((ext_vector_type(16))) __bf16 v16bf;
typedef __attribute__((ext_vector_type(8)))  float  v8f;
typedef __attribute__((ext_vector_type(4)))  float  v4f;
typedef __attribute__((ext_vector_type(4)))  int    v4i;

// Problem sizes from the reference (fixed).
constexpr int B = 8, L = 4096, H = 2048, E = 512, K = 64;
// softmax(scale*s): scale = 1/sqrt(K) = 0.125, folded into log2 domain.
constexpr float SCALE_LOG2E = 0.125f * 1.4426950408889634f;

#if __has_builtin(__builtin_amdgcn_global_load_async_to_lds_b128)
#define HAVE_ASYNC_LDS 1
#else
#define HAVE_ASYNC_LDS 0
#endif

DEVINL float fexp2(float x) { return __builtin_amdgcn_exp2f(x); }

DEVINL void wait_async0() {
#if HAVE_ASYNC_LDS
#if __has_builtin(__builtin_amdgcn_s_wait_asynccnt)
  __builtin_amdgcn_s_wait_asynccnt(0);
#else
  asm volatile("s_wait_asynccnt 0x0" ::: "memory");
#endif
#endif
}

// Copy NV4*256 float4's of contiguous global memory into contiguous LDS.
// Uses the CDNA5 async global->LDS DMA path (ASYNCcnt) when available:
// issue-only here, completion is claimed by wait_async0() at the consumer.
template <int NV4>
DEVINL void stage_block(float* lds, const float* glb, int tid) {
#if HAVE_ASYNC_LDS
  #pragma unroll
  for (int i = 0; i < NV4; ++i) {
    int o = (tid + i * 256) * 4;
    __builtin_amdgcn_global_load_async_to_lds_b128(
        (AS1 v4i*)(glb + o), (AS3 v4i*)(lds + o), 0, 0);
  }
#else
  #pragma unroll
  for (int i = 0; i < NV4; ++i) {
    int o = (tid + i * 256) * 4;
    *(v4f*)(lds + o) = *(const v4f*)(glb + o);
  }
#endif
}

DEVINL v8f wmma_bf16(v16bf a, v16bf b, v8f c) {
  // D = A(16x32 bf16) * B(32x16 bf16) + C(16x16 f32)
  return __builtin_amdgcn_wmma_f32_16x16x32_bf16(false, a, false, b,
                                                 (short)0, c, false, false);
}

DEVINL void cvt4(bf16* dst, v4f s) {
  dst[0] = (bf16)s.x; dst[1] = (bf16)s.y; dst[2] = (bf16)s.z; dst[3] = (bf16)s.w;
}

// Build A-operand (16x32 bf16) from an f32 LDS row (lane M = lane&15):
// 16 sequential K values, lanes>=16 take K+16.
// All four ds_load_b128 issued before the converts (single dscnt wait).
DEVINL v16bf make_a_op(const float* row, int lane, int kq) {
  const v4f* p = (const v4f*)(row + kq + ((lane < 16) ? 0 : 16));
  v4f c0 = p[0], c1 = p[1], c2 = p[2], c3 = p[3];
  v16bf a;
  cvt4((bf16*)&a + 0,  c0);
  cvt4((bf16*)&a + 4,  c1);
  cvt4((bf16*)&a + 8,  c2);
  cvt4((bf16*)&a + 12, c3);
  return a;
}

// Build B-operand (32x16 bf16) from an f32 LDS row (lane N = lane&15):
// 16 sequential K values, lanes>=16 take K+16 (64B contiguous).
DEVINL v16bf make_b_op(const float* row, int lane, int kq) {
  const v4f* p = (const v4f*)(row + kq + ((lane < 16) ? 0 : 16));
  v4f c0 = p[0], c1 = p[1], c2 = p[2], c3 = p[3];
  v16bf b;
  cvt4((bf16*)&b + 0,  c0);
  cvt4((bf16*)&b + 4,  c1);
  cvt4((bf16*)&b + 8,  c2);
  cvt4((bf16*)&b + 12, c3);
  return b;
}

// ---------------------------------------------------------------------------
// Stage 1: per-row softmax stats.  Grid (L/128, B), 256 threads (8 waves).
// Wave w owns 16 l-columns; loops over all H with WMMA S^T tiles and an
// online log2-domain max/sum.  Y chunks are double-buffered: the async DMA
// for chunk i+1 runs under the WMMA/exp work on chunk i.
// ---------------------------------------------------------------------------
__global__ __launch_bounds__(256) void t2v_stats_kernel(
    const float* __restrict__ xg, const float* __restrict__ yg,
    float* __restrict__ statM, float* __restrict__ statR) {
  __shared__ __align__(16) float Xblk[128][K];    // 32 KB
  __shared__ __align__(16) float Ybuf[2][64][K];  // 2 x 16 KB
  const int tid = threadIdx.x;
  const int lane = tid & 31, wave = tid >> 5;
  const int bb = blockIdx.y;
  const int lb = blockIdx.x * 128;
  const float* xp = xg + (size_t)bb * L * K;
  const float* yp = yg + (size_t)bb * H * K;

  // X block rows lb..lb+127 are contiguous (row stride == K): flat async copy
  stage_block<8>(&Xblk[0][0], xp + (size_t)lb * K, tid);
  stage_block<4>(&Ybuf[0][0][0], yp, tid);  // first Y chunk
  wait_async0();
  __syncthreads();

  // Per-wave B operands (X^T tile, reused across the whole H loop)
  const float* xrow = &Xblk[wave * 16 + (lane & 15)][0];
  const v16bf b0 = make_b_op(xrow, lane, 0);
  const v16bf b1 = make_b_op(xrow, lane, 32);

  float m = -1e30f, z = 0.0f;

  for (int hc = 0; hc < H; hc += 64) {
    const int cur = (hc >> 6) & 1;
    if (hc + 64 < H)  // prefetch next chunk under this chunk's compute
      stage_block<4>(&Ybuf[cur ^ 1][0][0], yp + (size_t)(hc + 64) * K, tid);

    #pragma unroll
    for (int ht = 0; ht < 4; ++ht) {
      const float* yrow = &Ybuf[cur][ht * 16 + (lane & 15)][0];
      v16bf a0 = make_a_op(yrow, lane, 0);
      v16bf a1 = make_a_op(yrow, lane, 32);
      v8f s = {};
      s = wmma_bf16(a0, b0, s);
      s = wmma_bf16(a1, b1, s);
      float t[8], cm = -1e30f;
      #pragma unroll
      for (int r = 0; r < 8; ++r) {
        t[r] = s[r] * SCALE_LOG2E;
        cm = fmaxf(cm, t[r]);
      }
      float mn = fmaxf(m, cm);
      float acc = z * fexp2(m - mn);
      #pragma unroll
      for (int r = 0; r < 8; ++r) acc += fexp2(t[r] - mn);
      z = acc; m = mn;
    }
    wait_async0();    // prefetch landed
    __syncthreads();  // all waves done with Ybuf[cur]
  }

  // Combine half-wave partials (same l column, complementary h rows)
  float om = __shfl_xor(m, 16, 32);
  float oz = __shfl_xor(z, 16, 32);
  float M = fmaxf(m, om);
  float Z = z * fexp2(m - M) + oz * fexp2(om - M);
  if (lane < 16) {
    int l = lb + wave * 16 + lane;
    statM[(size_t)bb * L + l] = M;
    statR[(size_t)bb * L + l] = 1.0f / Z;
  }
}

// ---------------------------------------------------------------------------
// Stage 2: Out[b, hb:hb+64, :] = sum_l P^T[h,l] * V[l,:].
// Grid (H/64, B), 256 threads (8 waves).  Wave w owns e-range [64w, 64w+64)
// across all 64 h rows: 4x4 f32 C tiles = 128 VGPRs/wave.
// Per 32-l chunk: the 8 S^T tiles are computed once cooperatively (one per
// wave), softmax-weighted into a shared LDS P buffer, then all waves reuse
// P^T as the A operand while streaming their own V columns from global.
// X chunks are double-buffered async copies.
// ---------------------------------------------------------------------------
__global__ __launch_bounds__(256) void t2v_attn_kernel(
    const float* __restrict__ vg, const float* __restrict__ xg,
    const float* __restrict__ yg, const float* __restrict__ statM,
    const float* __restrict__ statR, float* __restrict__ out) {
  __shared__ __align__(16) float Ybuf[64][K];      // 16 KB, staged once
  __shared__ __align__(16) float Xbuf[2][32][K];   // 2 x 8 KB
  __shared__ __align__(16) float Pbuf[64][32];     //  8 KB, shared P^T (f32)
  const int tid = threadIdx.x;
  const int lane = tid & 31, wave = tid >> 5;
  const int bb = blockIdx.y;
  const int hb = blockIdx.x * 64;
  const float* xp = xg + (size_t)bb * L * K;
  const float* yp = yg + (size_t)bb * H * K;
  const float* vp = vg + (size_t)bb * L * E;
  const float* mp = statM + (size_t)bb * L;
  const float* rp = statR + (size_t)bb * L;

  stage_block<4>(&Ybuf[0][0], yp + (size_t)hb * K, tid);
  stage_block<2>(&Xbuf[0][0][0], xp, tid);  // first X chunk
  wait_async0();
  __syncthreads();

  // Phase-A tile assignment: wave -> (h-tile, l-tile) of the 4x2 grid.
  const int ht = wave >> 1;
  const int lt = wave & 1;
  const float* yrow = &Ybuf[ht * 16 + (lane & 15)][0];
  const v16bf a1k0 = make_a_op(yrow, lane, 0);   // hoisted: Y fixed per block
  const v16bf a1k1 = make_a_op(yrow, lane, 32);

  v8f acc[4][4];
  const v8f vzero = {};
  #pragma unroll
  for (int i = 0; i < 4; ++i)
    #pragma unroll
    for (int j = 0; j < 4; ++j) acc[i][j] = vzero;

  for (int lb = 0; lb < L; lb += 32) {
    const int cur = (lb >> 5) & 1;
    if (lb + 32 < L)  // prefetch next X chunk under this iteration's work
      stage_block<2>(&Xbuf[cur ^ 1][0][0], xp + (size_t)(lb + 32) * K, tid);

    // Phase A: one S^T tile per wave, softmax-weight, publish to Pbuf.
    {
      const float* xrow = &Xbuf[cur][lt * 16 + (lane & 15)][0];
      v16bf bx0 = make_b_op(xrow, lane, 0);
      v16bf bx1 = make_b_op(xrow, lane, 32);
      v8f s = {};
      s = wmma_bf16(a1k0, bx0, s);
      s = wmma_bf16(a1k1, bx1, s);
      int l = lb + lt * 16 + (lane & 15);   // D lane column = l
      float m2 = mp[l], rz = rp[l];
      int hrow0 = ht * 16 + ((lane < 16) ? 0 : 8);  // D VGPR r -> M = r (+8)
      int lcol = lt * 16 + (lane & 15);
      #pragma unroll
      for (int r = 0; r < 8; ++r)
        Pbuf[hrow0 + r][lcol] = fexp2(fmaf(s[r], SCALE_LOG2E, -m2)) * rz;
    }
    __syncthreads();

    // Phase B: A2 = P^T tiles (shared by all waves), B2 = V columns.
    v16bf a2[4];
    #pragma unroll
    for (int t = 0; t < 4; ++t)
      a2[t] = make_a_op(&Pbuf[t * 16 + (lane & 15)][0], lane, 0);

    const int kb = lb + ((lane < 16) ? 0 : 16);
    #pragma unroll
    for (int u = 0; u < 4; ++u) {
      int e = wave * 64 + u * 16 + (lane & 15);
      // batch the 16 strided loads (64B/half-wave each), then convert
      float vt[16];
      #pragma unroll
      for (int j = 0; j < 16; ++j) vt[j] = vp[(size_t)(kb + j) * E + e];
      v16bf bv;
      #pragma unroll
      for (int j = 0; j < 16; ++j) ((bf16*)&bv)[j] = (bf16)vt[j];
      #pragma unroll
      for (int t = 0; t < 4; ++t) acc[t][u] = wmma_bf16(a2[t], bv, acc[t][u]);
    }
    wait_async0();    // next X chunk landed
    __syncthreads();  // all waves done with Pbuf / Xbuf[cur]
  }

  // Write Out[b][h][e]; C layout: VGPR r -> M = r (+8 for lanes>=16), N = lane&15.
  float* op = out + (size_t)bb * H * E;
  #pragma unroll
  for (int t = 0; t < 4; ++t) {
    int hrow0 = hb + t * 16 + ((lane < 16) ? 0 : 8);
    #pragma unroll
    for (int u = 0; u < 4; ++u) {
      int e = wave * 64 + u * 16 + (lane & 15);
      #pragma unroll
      for (int r = 0; r < 8; ++r)
        op[(size_t)(hrow0 + r) * E + e] = acc[t][u][r];
    }
  }
}

extern "C" void kernel_launch(void* const* d_in, const int* in_sizes, int n_in,
                              void* d_out, int out_size, void* d_ws, size_t ws_size,
                              hipStream_t stream) {
  (void)in_sizes; (void)n_in; (void)out_size; (void)ws_size;
  // setup_inputs order: queries(0), keys(1) unused; values(2), x_date(3), y_date(4)
  const float* values = (const float*)d_in[2];
  const float* x_date = (const float*)d_in[3];
  const float* y_date = (const float*)d_in[4];
  float* out = (float*)d_out;
  // Workspace: softmax row stats, 2 * B * L floats = 256 KB.
  float* statM = (float*)d_ws;
  float* statR = statM + (size_t)B * L;

  t2v_stats_kernel<<<dim3(L / 128, B), dim3(256), 0, stream>>>(
      x_date, y_date, statM, statR);
  t2v_attn_kernel<<<dim3(H / 64, B), dim3(256), 0, stream>>>(
      values, x_date, y_date, statM, statR, out);
}
